// LBANP_2284922601866
// MI455X (gfx1250) — matrix operand order repaired and checked by reading, not compiled, not run on hardware
//
#include <hip/hip_runtime.h>

// ---------------- model dims ----------------
#define B_   16
#define NC_  2048
#define NT_  2048
#define DX_  32
#define DY_  16
#define D_   512
#define H_   8
#define DFF_ 2048
#define NL_  128
#define DH_  64

// ---------------- types ----------------
typedef __bf16 bf16;
typedef __bf16 v16bf __attribute__((ext_vector_type(16)));
typedef __bf16 v8bf  __attribute__((ext_vector_type(8)));
typedef float  v8f   __attribute__((ext_vector_type(8)));

union FragA { v16bf v; v8bf h[2]; };

__device__ __forceinline__ bf16 f2bf(float f) { return (bf16)f; }  // native v_cvt path
__device__ __forceinline__ v8bf pack8(float4 a, float4 b) {
  v8bf r;
  r[0] = f2bf(a.x); r[1] = f2bf(a.y); r[2] = f2bf(a.z); r[3] = f2bf(a.w);
  r[4] = f2bf(b.x); r[5] = f2bf(b.y); r[6] = f2bf(b.z); r[7] = f2bf(b.w);
  return r;
}

// ============================================================
// Generic batched GEMM: C = act(alpha * A@B + bias) (+ resid)
// A[m,k] = A[aoff + m*lda + k]           (row-major, k contiguous)
// B^T[n,k] = B[boff + n*ldbn + k*ldbk]
// C[m,n] = C[coff + m*ldc + n]
// Block tile 128x64, 8 waves, wave tile 32x32 (4 WMMAs per K=32 step).
// Double-buffered LDS: fill(i+1) overlaps WMMA(i); one barrier per step.
// z-batch offsets: off = (z/hdiv)*bs + (z%hdiv)*hs.  act: 0=none, 1=relu
// GUARD=false: M%128==0, N%64==0, K%32==0 (no bounds checks at all).
// BKC: B is k-contiguous (ldbk==1) -> row loads; else n-contiguous scatter.
// ============================================================
template <bool GUARD, bool BKC>
__global__ __launch_bounds__(256) void gemm_wmma(
    const float* __restrict__ A, const float* __restrict__ Bm,
    const float* __restrict__ bias, const float* __restrict__ resid,
    float* __restrict__ C,
    int M, int N, int K,
    int lda, int ldbn, int ldbk, int ldc,
    long a_bs, long a_hs, long b_bs, long b_hs, long c_bs, long c_hs,
    int hdiv, float alpha, int act)
{
  __shared__ __align__(64) bf16 As[2][128][32];
  __shared__ __align__(64) bf16 Bs[2][64][32];

  const int z  = blockIdx.z;
  const int zb = z / hdiv, zh = z % hdiv;
  const long aoff = (long)zb * a_bs + (long)zh * a_hs;
  const long boff = (long)zb * b_bs + (long)zh * b_hs;
  const long coff = (long)zb * c_bs + (long)zh * c_hs;

  const int m0 = blockIdx.y * 128;
  const int n0 = blockIdx.x * 64;
  const int tid = threadIdx.x;
  const int lane = tid & 31, wv = tid >> 5;
  const int wm = wv >> 1, wn = wv & 1;   // wave grid 4 (M) x 2 (N); wave tile 32x32

  v8f c00 = {0.f,0.f,0.f,0.f,0.f,0.f,0.f,0.f};
  v8f c01 = c00, c10 = c00, c11 = c00;

  // A fill: 128 rows x 32 cols, thread -> 1 row, 16 cols
  const int arow = tid >> 1;
  const int acb  = (tid & 1) << 4;
  // B fill (k-contiguous): 64 rows x 32 cols, thread -> 1 row, 8 cols
  const int brow = tid >> 2;
  const int bcb  = (tid & 3) << 3;
  // B fill (n-contiguous): thread -> 1 k, 8 n values
  const int bkk = tid >> 3;
  const int bnb = (tid & 7) << 3;

  auto fill = [&](int st, int k0) {
    // ---- A tile ----
    {
      const long gm = m0 + arow;
      if (!GUARD || (gm < M && k0 + 32 <= K)) {
        const float4* p = (const float4*)(A + aoff + gm * (long)lda + k0 + acb);
        float4 x0 = p[0], x1 = p[1], x2 = p[2], x3 = p[3];
        *(v8bf*)&As[st][arow][acb]     = pack8(x0, x1);
        *(v8bf*)&As[st][arow][acb + 8] = pack8(x2, x3);
      } else {
#pragma unroll
        for (int j = 0; j < 16; ++j) {
          const int  kk = acb + j;
          const long gk = k0 + kk;
          float v = 0.f;
          if (gm < M && gk < K) v = A[aoff + gm * (long)lda + gk];
          As[st][arow][kk] = f2bf(v);
        }
      }
    }
    // ---- B tile ----
    if (BKC) {
      const long gn = n0 + brow;
      if (!GUARD || (gn < N && k0 + 32 <= K)) {
        const float4* p = (const float4*)(Bm + boff + gn * (long)ldbn + k0 + bcb);
        *(v8bf*)&Bs[st][brow][bcb] = pack8(p[0], p[1]);
      } else {
#pragma unroll
        for (int j = 0; j < 8; ++j) {
          const int  kk = bcb + j;
          const long gk = k0 + kk;
          float v = 0.f;
          if (gn < N && gk < K) v = Bm[boff + gn * (long)ldbn + gk];
          Bs[st][brow][kk] = f2bf(v);
        }
      }
    } else {
      const long gk = k0 + bkk;
      const long gn = n0 + bnb;
      if (!GUARD || (gk < K && gn + 7 < N)) {
        const float4* p = (const float4*)(Bm + boff + gk * (long)ldbk + gn * (long)ldbn);
        float4 x0 = p[0], x1 = p[1];
        Bs[st][bnb + 0][bkk] = f2bf(x0.x); Bs[st][bnb + 1][bkk] = f2bf(x0.y);
        Bs[st][bnb + 2][bkk] = f2bf(x0.z); Bs[st][bnb + 3][bkk] = f2bf(x0.w);
        Bs[st][bnb + 4][bkk] = f2bf(x1.x); Bs[st][bnb + 5][bkk] = f2bf(x1.y);
        Bs[st][bnb + 6][bkk] = f2bf(x1.z); Bs[st][bnb + 7][bkk] = f2bf(x1.w);
      } else {
#pragma unroll
        for (int j = 0; j < 8; ++j) {
          const long gnj = gn + j;
          float v = 0.f;
          if (gk < K && gnj < N) v = Bm[boff + gk * (long)ldbk + gnj * (long)ldbn];
          Bs[st][bnb + j][bkk] = f2bf(v);
        }
      }
    }
  };

  const int ln15 = lane & 15;
  const int kh = (lane >> 4) << 4;   // A: contiguous 16 K values
  const int kb = (lane >> 4) << 4;   // B: contiguous 16 K values
  const int ra0 = wm * 32 + ln15;
  const int rb  = wn * 32 + ln15;

  auto compute = [&](int st) {
    FragA fa0, fa1;
    fa0.h[0] = *(const v8bf*)&As[st][ra0][kh];
    fa0.h[1] = *(const v8bf*)&As[st][ra0][kh + 8];
    fa1.h[0] = *(const v8bf*)&As[st][ra0 + 16][kh];
    fa1.h[1] = *(const v8bf*)&As[st][ra0 + 16][kh + 8];
    const v16bf fb0 = *(const v16bf*)&Bs[st][rb][kb];
    const v16bf fb1 = *(const v16bf*)&Bs[st][rb + 16][kb];
    c00 = __builtin_amdgcn_wmma_f32_16x16x32_bf16(false, fa0.v, false, fb0, (short)0, c00, false, false);
    c01 = __builtin_amdgcn_wmma_f32_16x16x32_bf16(false, fa0.v, false, fb1, (short)0, c01, false, false);
    c10 = __builtin_amdgcn_wmma_f32_16x16x32_bf16(false, fa1.v, false, fb0, (short)0, c10, false, false);
    c11 = __builtin_amdgcn_wmma_f32_16x16x32_bf16(false, fa1.v, false, fb1, (short)0, c11, false, false);
  };

  const int nsteps = GUARD ? (K + 31) / 32 : (K >> 5);
  fill(0, 0);
  __syncthreads();
  for (int i = 0; i < nsteps; ++i) {
    const int cur = i & 1;
    if (i + 1 < nsteps) fill(cur ^ 1, (i + 1) << 5);
    compute(cur);
    __syncthreads();
  }

  // ---- epilogue ----
  const int mb  = m0 + wm * 32 + ((lane >> 4) << 3);
  const int nb0 = n0 + wn * 32 + ln15;
#pragma unroll
  for (int r = 0; r < 8; ++r) {
    const int gm0 = mb + r;
    const int gm1 = gm0 + 16;
    const int gnA = nb0;
    const int gnB = nb0 + 16;
    if (!GUARD || gm0 < M) {
      const long crow = coff + (long)gm0 * ldc;
      if (!GUARD || gnA < N) {
        float v = c00[r] * alpha + (bias ? bias[gnA] : 0.f);
        if (act == 1) v = fmaxf(v, 0.f);
        if (resid) v += resid[crow + gnA];
        C[crow + gnA] = v;
      }
      if (!GUARD || gnB < N) {
        float v = c01[r] * alpha + (bias ? bias[gnB] : 0.f);
        if (act == 1) v = fmaxf(v, 0.f);
        if (resid) v += resid[crow + gnB];
        C[crow + gnB] = v;
      }
    }
    if (!GUARD || gm1 < M) {
      const long crow = coff + (long)gm1 * ldc;
      if (!GUARD || gnA < N) {
        float v = c10[r] * alpha + (bias ? bias[gnA] : 0.f);
        if (act == 1) v = fmaxf(v, 0.f);
        if (resid) v += resid[crow + gnA];
        C[crow + gnA] = v;
      }
      if (!GUARD || gnB < N) {
        float v = c11[r] * alpha + (bias ? bias[gnB] : 0.f);
        if (act == 1) v = fmaxf(v, 0.f);
        if (resid) v += resid[crow + gnB];
        C[crow + gnB] = v;
      }
    }
  }
}

// ============================================================
// GLU GEMM: C[m,n] = (A@W[:,n]+b1[n]) * gelu(A@W[:,n+NFF]+b1[n+NFF])
// W is [K, 2*NFF]; hidden [M,2*NFF] never materialized.
// Shapes always exact here: M%64==0, K%32==0, NFF%64==0 -> no guards.
// Block 64x64, wave tile 16x32 over two B tiles (4 WMMAs/step).
// Double-buffered LDS pipeline.
// ============================================================
__global__ __launch_bounds__(256) void gemm_glu_wmma(
    const float* __restrict__ A, const float* __restrict__ W,
    const float* __restrict__ b1, float* __restrict__ C,
    int M, int K, int NFF)
{
  __shared__ __align__(64) bf16 As[2][64][32];
  __shared__ __align__(64) bf16 Ba[2][64][32];
  __shared__ __align__(64) bf16 Bg[2][64][32];

  const int ldw = 2 * NFF;
  const int m0 = blockIdx.y * 64;
  const int n0 = blockIdx.x * 64;
  const int tid = threadIdx.x;
  const int lane = tid & 31, wv = tid >> 5;
  const int wm = wv >> 1, wn = wv & 1;

  v8f zro = {0.f,0.f,0.f,0.f,0.f,0.f,0.f,0.f};
  v8f aa0 = zro, aa1 = zro, ag0 = zro, ag1 = zro;

  const int arow = tid >> 2;          // A: 64 rows, 8 cols/thread
  const int acb  = (tid & 3) << 3;
  const int bkk  = tid >> 3;          // W: 1 k, 8 n/thread
  const int bnb  = (tid & 7) << 3;

  auto fill = [&](int st, int k0) {
    {
      const long gm = m0 + arow;
      const float4* p = (const float4*)(A + gm * (long)K + k0 + acb);
      *(v8bf*)&As[st][arow][acb] = pack8(p[0], p[1]);
    }
    {
      const long gk = k0 + bkk;
      const long gn = n0 + bnb;
      const float4* pa = (const float4*)(W + gk * (long)ldw + gn);
      const float4* pg = (const float4*)(W + gk * (long)ldw + gn + NFF);
      float4 a0 = pa[0], a1 = pa[1], g0 = pg[0], g1 = pg[1];
      Ba[st][bnb + 0][bkk] = f2bf(a0.x); Ba[st][bnb + 1][bkk] = f2bf(a0.y);
      Ba[st][bnb + 2][bkk] = f2bf(a0.z); Ba[st][bnb + 3][bkk] = f2bf(a0.w);
      Ba[st][bnb + 4][bkk] = f2bf(a1.x); Ba[st][bnb + 5][bkk] = f2bf(a1.y);
      Ba[st][bnb + 6][bkk] = f2bf(a1.z); Ba[st][bnb + 7][bkk] = f2bf(a1.w);
      Bg[st][bnb + 0][bkk] = f2bf(g0.x); Bg[st][bnb + 1][bkk] = f2bf(g0.y);
      Bg[st][bnb + 2][bkk] = f2bf(g0.z); Bg[st][bnb + 3][bkk] = f2bf(g0.w);
      Bg[st][bnb + 4][bkk] = f2bf(g1.x); Bg[st][bnb + 5][bkk] = f2bf(g1.y);
      Bg[st][bnb + 6][bkk] = f2bf(g1.z); Bg[st][bnb + 7][bkk] = f2bf(g1.w);
    }
  };

  const int ln15 = lane & 15;
  const int kh = (lane >> 4) << 4;
  const int kb = (lane >> 4) << 4;
  const int ra = wm * 16 + ln15;
  const int rb = wn * 32 + ln15;

  auto compute = [&](int st) {
    FragA fa;
    fa.h[0] = *(const v8bf*)&As[st][ra][kh];
    fa.h[1] = *(const v8bf*)&As[st][ra][kh + 8];
    const v16bf ba0 = *(const v16bf*)&Ba[st][rb][kb];
    const v16bf ba1 = *(const v16bf*)&Ba[st][rb + 16][kb];
    const v16bf bg0 = *(const v16bf*)&Bg[st][rb][kb];
    const v16bf bg1 = *(const v16bf*)&Bg[st][rb + 16][kb];
    aa0 = __builtin_amdgcn_wmma_f32_16x16x32_bf16(false, fa.v, false, ba0, (short)0, aa0, false, false);
    aa1 = __builtin_amdgcn_wmma_f32_16x16x32_bf16(false, fa.v, false, ba1, (short)0, aa1, false, false);
    ag0 = __builtin_amdgcn_wmma_f32_16x16x32_bf16(false, fa.v, false, bg0, (short)0, ag0, false, false);
    ag1 = __builtin_amdgcn_wmma_f32_16x16x32_bf16(false, fa.v, false, bg1, (short)0, ag1, false, false);
  };

  const int nsteps = K >> 5;
  fill(0, 0);
  __syncthreads();
  for (int i = 0; i < nsteps; ++i) {
    const int cur = i & 1;
    if (i + 1 < nsteps) fill(cur ^ 1, (i + 1) << 5);
    compute(cur);
    __syncthreads();
  }

  const int mbase = m0 + wm * 16 + ((lane >> 4) << 3);
  const int gn0 = n0 + wn * 32 + ln15;
  const int gn1 = gn0 + 16;
#pragma unroll
  for (int r = 0; r < 8; ++r) {
    const int gm = mbase + r;
    {
      const float a = aa0[r] + b1[gn0];
      const float g = ag0[r] + b1[gn0 + NFF];
      C[(long)gm * NFF + gn0] = a * 0.5f * g * (1.f + erff(g * 0.70710678118654752f));
    }
    {
      const float a = aa1[r] + b1[gn1];
      const float g = ag1[r] + b1[gn1 + NFF];
      C[(long)gm * NFF + gn1] = a * 0.5f * g * (1.f + erff(g * 0.70710678118654752f));
    }
  }
}

// ============================================================
// LayerNorm over rows of length D_=512: one wave per row.
// ============================================================
__global__ __launch_bounds__(256) void layernorm_rows(
    const float* __restrict__ X, const float* __restrict__ g,
    const float* __restrict__ b, float* __restrict__ Y, int rows)
{
  const int wv = threadIdx.x >> 5, lane = threadIdx.x & 31;
  const int row = blockIdx.x * 8 + wv;
  if (row >= rows) return;
  const float* x = X + (long)row * D_;
  float v[16];
  float s = 0.f;
#pragma unroll
  for (int i = 0; i < 16; ++i) { v[i] = x[lane + i * 32]; s += v[i]; }
#pragma unroll
  for (int o = 16; o > 0; o >>= 1) s += __shfl_xor(s, o, 32);
  const float mean = s * (1.f / (float)D_);
  float var = 0.f;
#pragma unroll
  for (int i = 0; i < 16; ++i) { const float d = v[i] - mean; var += d * d; }
#pragma unroll
  for (int o = 16; o > 0; o >>= 1) var += __shfl_xor(var, o, 32);
  const float inv = rsqrtf(var * (1.f / (float)D_) + 1e-5f);
  float* y = Y + (long)row * D_;
#pragma unroll
  for (int i = 0; i < 16; ++i) {
    const int c = lane + i * 32;
    y[c] = (v[i] - mean) * inv * g[c] + b[c];
  }
}

// ============================================================
// Softmax over last dim, one wave per row.
// ============================================================
__global__ __launch_bounds__(256) void softmax_rows(float* __restrict__ S, long rows, int ncols)
{
  const int wv = threadIdx.x >> 5, lane = threadIdx.x & 31;
  const long row = (long)blockIdx.x * 8 + wv;
  if (row >= rows) return;
  float* s = S + row * (long)ncols;
  float mx = -3.402823466e38f;
  for (int c = lane; c < ncols; c += 32) mx = fmaxf(mx, s[c]);
  for (int o = 16; o > 0; o >>= 1) mx = fmaxf(mx, __shfl_xor(mx, o, 32));
  float sum = 0.f;
  for (int c = lane; c < ncols; c += 32) { const float e = __expf(s[c] - mx); s[c] = e; sum += e; }
  for (int o = 16; o > 0; o >>= 1) sum += __shfl_xor(sum, o, 32);
  const float inv = 1.f / sum;
  for (int c = lane; c < ncols; c += 32) s[c] *= inv;
}

// ---------------- elementwise helpers ----------------
__global__ void concat_xy(const float* __restrict__ xc, const float* __restrict__ yc,
                          float* __restrict__ cat, long n)
{
  const long i = (long)blockIdx.x * blockDim.x + threadIdx.x;
  if (i >= n) return;
  const long j = i % 48, t = i / 48;
  cat[i] = (j < DX_) ? xc[t * DX_ + j] : yc[t * DY_ + (j - DX_)];
}
__global__ void bcast_latents(const float* __restrict__ lat, float* __restrict__ x, long n)
{
  const long i = (long)blockIdx.x * blockDim.x + threadIdx.x;
  if (i < n) x[i] = lat[i % ((long)NL_ * D_)];
}
__global__ void copy_f(const float* __restrict__ s, float* __restrict__ d, long n)
{
  const long i = (long)blockIdx.x * blockDim.x + threadIdx.x;
  if (i < n) d[i] = s[i];
}

// ================= host orchestration =================
namespace {

struct AttnP { const float *ln_q_g,*ln_q_b,*Wq,*Wkv,*Wo,*bo,*ln_c_g,*ln_c_b; };
struct FFP   { const float *ln_g,*ln_b,*W1,*b1,*W2,*b2; };

inline void gemm(hipStream_t st, const float* A, const float* Bm, const float* bias,
                 const float* resid, float* C, int M, int N, int K,
                 int lda, int ldbn, int ldbk, int ldc,
                 long a_bs, long a_hs, long b_bs, long b_hs, long c_bs, long c_hs,
                 int batches, int hdiv, float alpha, int act)
{
  dim3 grid((unsigned)((N + 63) / 64), (unsigned)((M + 127) / 128), (unsigned)batches);
  const bool guard = (M % 128) || (N % 64) || (K % 32);
  const bool bkc = (ldbk == 1);
  if (guard) {
    if (bkc) gemm_wmma<true, true><<<grid, dim3(256), 0, st>>>(A, Bm, bias, resid, C, M, N, K,
        lda, ldbn, ldbk, ldc, a_bs, a_hs, b_bs, b_hs, c_bs, c_hs, hdiv, alpha, act);
    else     gemm_wmma<true, false><<<grid, dim3(256), 0, st>>>(A, Bm, bias, resid, C, M, N, K,
        lda, ldbn, ldbk, ldc, a_bs, a_hs, b_bs, b_hs, c_bs, c_hs, hdiv, alpha, act);
  } else {
    if (bkc) gemm_wmma<false, true><<<grid, dim3(256), 0, st>>>(A, Bm, bias, resid, C, M, N, K,
        lda, ldbn, ldbk, ldc, a_bs, a_hs, b_bs, b_hs, c_bs, c_hs, hdiv, alpha, act);
    else     gemm_wmma<false, false><<<grid, dim3(256), 0, st>>>(A, Bm, bias, resid, C, M, N, K,
        lda, ldbn, ldbk, ldc, a_bs, a_hs, b_bs, b_hs, c_bs, c_hs, hdiv, alpha, act);
  }
}
inline void gemm2d(hipStream_t st, const float* A, const float* W, const float* bias,
                   const float* resid, float* C, int M, int N, int K, int act)
{
  gemm(st, A, W, bias, resid, C, M, N, K, K, 1, N, N,
       0, 0, 0, 0, 0, 0, 1, 1, 1.f, act);
}
inline void lnorm(hipStream_t st, const float* X, const float* g, const float* b,
                  float* Y, long rows)
{
  layernorm_rows<<<dim3((unsigned)((rows + 7) / 8)), dim3(256), 0, st>>>(X, g, b, Y, (int)rows);
}

// state: [B_, Mtok, D_]; ctxb: [B_, Ntok, D_] or nullptr (self-attention)
inline void attn_block(hipStream_t st, float* XN, float* CN, float* QB, float* KVB,
                       float* SIM, float* OB, float* state, const float* ctxb,
                       int Mtok, int Ntok, const AttnP& p)
{
  const long rowsQ = (long)B_ * Mtok;
  const long rowsK = (long)B_ * Ntok;
  lnorm(st, state, p.ln_q_g, p.ln_q_b, XN, rowsQ);
  const float* kvsrc = XN;
  if (ctxb) { lnorm(st, ctxb, p.ln_c_g, p.ln_c_b, CN, rowsK); kvsrc = CN; }
  gemm2d(st, XN, p.Wq, nullptr, nullptr, QB, (int)rowsQ, D_, D_, 0);
  gemm2d(st, kvsrc, p.Wkv, nullptr, nullptr, KVB, (int)rowsK, 2 * D_, D_, 0);
  // sim[b,h,i,j] = (1/8) * q . k   (B operand = K rows, k-contiguous)
  gemm(st, QB, KVB, nullptr, nullptr, SIM,
       Mtok, Ntok, DH_,
       D_, 2 * D_, 1, Ntok,
       (long)Mtok * D_, DH_,
       (long)Ntok * 2 * D_, DH_,
       (long)H_ * Mtok * Ntok, (long)Mtok * Ntok,
       B_ * H_, H_, 0.125f, 0);
  const long srows = (long)B_ * H_ * Mtok;
  softmax_rows<<<dim3((unsigned)((srows + 7) / 8)), dim3(256), 0, st>>>(SIM, srows, Ntok);
  // o[b,i,h*DH+d] = sum_j a[b,h,i,j] * v[b,j,h*DH+d]   (v = KVB column offset +D_)
  gemm(st, SIM, KVB + D_, nullptr, nullptr, OB,
       Mtok, DH_, Ntok,
       Ntok, 1, 2 * D_, D_,
       (long)H_ * Mtok * Ntok, (long)Mtok * Ntok,
       (long)Ntok * 2 * D_, DH_,
       (long)Mtok * D_, DH_,
       B_ * H_, H_, 1.f, 0);
  // out = o @ Wo + bo + state  (residual), in-place into state
  gemm2d(st, OB, p.Wo, p.bo, state, state, (int)rowsQ, D_, D_, 0);
}

inline void ff_block(hipStream_t st, float* XN, float* HID, float* state, long rows, const FFP& p)
{
  lnorm(st, state, p.ln_g, p.ln_b, XN, rows);
  dim3 grid((unsigned)(DFF_ / 64), (unsigned)((rows + 63) / 64), 1);
  gemm_glu_wmma<<<grid, dim3(256), 0, st>>>(XN, p.W1, p.b1, HID, (int)rows, D_, DFF_);
  gemm2d(st, HID, p.W2, p.b2, state, state, (int)rows, D_, DFF_, 0);
}

} // namespace

extern "C" void kernel_launch(void* const* d_in, const int* in_sizes, int n_in,
                              void* d_out, int out_size, void* d_ws, size_t ws_size,
                              hipStream_t stream)
{
  (void)in_sizes; (void)n_in; (void)out_size; (void)ws_size;
  int idx = 0;
  auto IN = [&]() -> const float* { return (const float*)d_in[idx++]; };

  const float* xc = IN();
  const float* yc = IN();
  const float* xt = IN();
  const float* latents = IN();

  const float *embW[4], *embB[4];
  for (int i = 0; i < 4; ++i) { embW[i] = IN(); embB[i] = IN(); }
  const float *qembW[4], *qembB[4];
  for (int i = 0; i < 4; ++i) { qembW[i] = IN(); qembB[i] = IN(); }

  AttnP enc_ca[6], enc_sa[6];
  FFP   enc_cff[6], enc_sff[6];
  for (int l = 0; l < 6; ++l) {
    AttnP a; a.ln_q_g=IN(); a.ln_q_b=IN(); a.Wq=IN(); a.Wkv=IN(); a.Wo=IN(); a.bo=IN();
    a.ln_c_g=IN(); a.ln_c_b=IN(); enc_ca[l]=a;
    FFP f; f.ln_g=IN(); f.ln_b=IN(); f.W1=IN(); f.b1=IN(); f.W2=IN(); f.b2=IN(); enc_cff[l]=f;
    AttnP s; s.ln_q_g=IN(); s.ln_q_b=IN(); s.Wq=IN(); s.Wkv=IN(); s.Wo=IN(); s.bo=IN();
    s.ln_c_g=nullptr; s.ln_c_b=nullptr; enc_sa[l]=s;
    FFP f2; f2.ln_g=IN(); f2.ln_b=IN(); f2.W1=IN(); f2.b1=IN(); f2.W2=IN(); f2.b2=IN(); enc_sff[l]=f2;
  }
  AttnP dec_ca[6];
  FFP   dec_cff[6];
  for (int l = 0; l < 6; ++l) {
    AttnP a; a.ln_q_g=IN(); a.ln_q_b=IN(); a.Wq=IN(); a.Wkv=IN(); a.Wo=IN(); a.bo=IN();
    a.ln_c_g=IN(); a.ln_c_b=IN(); dec_ca[l]=a;
    FFP f; f.ln_g=IN(); f.ln_b=IN(); f.W1=IN(); f.b1=IN(); f.W2=IN(); f.b2=IN(); dec_cff[l]=f;
  }
  const float* norm_g = IN();
  const float* norm_b = IN();
  const float* pW1 = IN(); const float* pb1 = IN();
  const float* pW2 = IN(); const float* pb2 = IN();

  // ---------------- workspace layout (floats) ----------------
  float* ws = (float*)d_ws;
  long o = 0;
  float* CAT = ws + o; o += (long)B_ * NC_ * 48;
  float* CTX = ws + o; o += (long)B_ * NC_ * D_;
  float* Y   = ws + o; o += (long)B_ * NT_ * D_;      // decoder state (query embedding)
  float* XB  = ws + o; o += (long)B_ * NL_ * D_;      // latent state
  float* ENC = ws + o; o += 6L * B_ * NL_ * D_;       // encoder outputs
  float* XN  = ws + o; o += (long)B_ * NT_ * D_;      // normed queries / ping-pong
  float* CN  = ws + o; o += (long)B_ * NC_ * D_;      // normed context
  float* QB  = ws + o; o += (long)B_ * NT_ * D_;      // q projection
  float* OB  = ws + o; o += (long)B_ * NT_ * D_;      // attention output
  float* KVB = ws + o; o += (long)B_ * NC_ * 2 * D_;  // kv projection
  float* SIM = ws + o; o += (long)B_ * H_ * (long)NT_ * NL_;  // attention scores
  float* HID = KVB;  // FFN hidden [B*NT, DFF] aliases (KVB+SIM), both dead during FFN

  // 1) concat(xc, yc) -> CAT [B*NC, 48]
  { const long n = (long)B_ * NC_ * 48;
    concat_xy<<<dim3((unsigned)((n + 255) / 256)), dim3(256), 0, stream>>>(xc, yc, CAT, n); }

  // 2) embedder MLP -> CTX  (ping-pong through XN/OB)
  gemm2d(stream, CAT, embW[0], embB[0], nullptr, XN,  B_ * NC_, D_, DX_ + DY_, 1);
  gemm2d(stream, XN,  embW[1], embB[1], nullptr, OB,  B_ * NC_, D_, D_, 1);
  gemm2d(stream, OB,  embW[2], embB[2], nullptr, XN,  B_ * NC_, D_, D_, 1);
  gemm2d(stream, XN,  embW[3], embB[3], nullptr, CTX, B_ * NC_, D_, D_, 0);

  // 3) query embedder MLP -> Y
  gemm2d(stream, xt, qembW[0], qembB[0], nullptr, XN, B_ * NT_, D_, DX_, 1);
  gemm2d(stream, XN, qembW[1], qembB[1], nullptr, OB, B_ * NT_, D_, D_, 1);
  gemm2d(stream, OB, qembW[2], qembB[2], nullptr, XN, B_ * NT_, D_, D_, 1);
  gemm2d(stream, XN, qembW[3], qembB[3], nullptr, Y,  B_ * NT_, D_, D_, 0);

  // 4) broadcast latents -> XB
  { const long n = (long)B_ * NL_ * D_;
    bcast_latents<<<dim3((unsigned)((n + 255) / 256)), dim3(256), 0, stream>>>(latents, XB, n); }

  // 5) encoder
  for (int l = 0; l < 6; ++l) {
    attn_block(stream, XN, CN, QB, KVB, SIM, OB, XB, CTX,     NL_, NC_, enc_ca[l]);
    ff_block  (stream, XN, HID, XB, (long)B_ * NL_, enc_cff[l]);
    attn_block(stream, XN, CN, QB, KVB, SIM, OB, XB, nullptr, NL_, NL_, enc_sa[l]);
    ff_block  (stream, XN, HID, XB, (long)B_ * NL_, enc_sff[l]);
    const long n = (long)B_ * NL_ * D_;
    copy_f<<<dim3((unsigned)((n + 255) / 256)), dim3(256), 0, stream>>>(XB, ENC + (long)l * n, n);
  }

  // 6) decoder
  for (int l = 0; l < 6; ++l) {
    attn_block(stream, XN, CN, QB, KVB, SIM, OB, Y, ENC + (long)l * B_ * NL_ * D_,
               NT_, NL_, dec_ca[l]);
    ff_block(stream, XN, HID, Y, (long)B_ * NT_, dec_cff[l]);
  }

  // 7) final norm + prediction head
  lnorm(stream, Y, norm_g, norm_b, XN, (long)B_ * NT_);
  gemm2d(stream, XN,  pW1, pb1, nullptr, HID,            B_ * NT_, DFF_,     D_,   1);
  gemm2d(stream, HID, pW2, pb2, nullptr, (float*)d_out,  B_ * NT_, 2 * DY_,  DFF_, 0);
}
